// QuantizedLinear_4964982194484
// MI455X (gfx1250) — 8-bit matrix operand order repaired and checked by reading, compiled, not run on hardware
//
#include <hip/hip_runtime.h>
#include <hip/hip_bf16.h>
#include <stdint.h>

#define BATCH_T   8192
#define KDIM      4096
#define KN        3968
#define KKSTEPS   62
#define KO        128
#define NDIM      4096

#define XQ_OFF    0
#define XO_OFF    (BATCH_T * KN)
#define XS_OFF    (XO_OFF + BATCH_T * KO * 4)

typedef __attribute__((ext_vector_type(8)))  int   v8i;
typedef __attribute__((ext_vector_type(4)))  int   v4i;
typedef __attribute__((ext_vector_type(8)))  float v8f;
typedef __attribute__((ext_vector_type(2)))  float v2f;

__global__ void __launch_bounds__(256)
quant_gather_kernel(const float* __restrict__ x,
                    const int* __restrict__ nidx,
                    const int* __restrict__ oidx,
                    int8_t* __restrict__ xqs,
                    float* __restrict__ xo,
                    float* __restrict__ xs) {
    const int t   = blockIdx.x;
    const int tid = threadIdx.x;
    const float* xrow = x + (size_t)t * KDIM;

    __shared__ float s_red[256];
    __shared__ float s_scale;

    float amax = 0.0f;
    for (int k = tid; k < KN; k += 256) {
        float v = xrow[nidx[k]];
        amax = fmaxf(amax, fabsf(v));
    }
    s_red[tid] = amax;
    __syncthreads();
    for (int s = 128; s > 0; s >>= 1) {
        if (tid < s) s_red[tid] = fmaxf(s_red[tid], s_red[tid + s]);
        __syncthreads();
    }
    if (tid == 0) {
        float sc = fmaxf(s_red[0], 1e-8f) / 127.0f;
        s_scale = sc;
        xs[t] = sc;
    }
    __syncthreads();
    const float sc = s_scale;

    const int tileM = t >> 4;
    const int m     = t & 15;
    int8_t* frag_base = xqs + (size_t)tileM * KKSTEPS * 1024;

    for (int k = tid; k < KN; k += 256) {
        float v = xrow[nidx[k]];
        float q = rintf(v / sc);
        q = fminf(127.0f, fmaxf(-127.0f, q));
        const int kk   = k >> 6;
        const int r    = k & 63;
        const int hi   = (r >> 4) & 1;
        const int lane = m + (hi << 4);
        const int pos  = ((r >> 5) << 4) + (r & 15);
        frag_base[kk * 1024 + lane * 32 + pos] = (int8_t)q;
    }

    if (tid < KO) {
        xo[(size_t)t * KO + tid] = xrow[oidx[tid]];
    }
}

static __device__ __forceinline__ v8i pack8(v4i lo, v4i hi8) {
    v8i r;
    r[0] = lo[0]; r[1] = lo[1]; r[2] = lo[2]; r[3] = lo[3];
    r[4] = hi8[0]; r[5] = hi8[1]; r[6] = hi8[2]; r[7] = hi8[3];
    return r;
}

__global__ void __launch_bounds__(256)
wmma_gemm_kernel(const int8_t* __restrict__ xqs,
                 const float* __restrict__ xs,
                 const int8_t* __restrict__ w8,
                 const float* __restrict__ wsc,
                 const float* __restrict__ xo,
                 const float* __restrict__ wo,
                 float* __restrict__ out) {
    const int lane = threadIdx.x & 31;
    const int wid  = threadIdx.x >> 5;
    const int tileM = blockIdx.x >> 3;
    const int n4    = ((blockIdx.x & 7) << 3) + wid;
    const int colBase = n4 << 6;

    const int l15 = lane & 15;
    const int hi  = lane >> 4;

    const int8_t* aptr = xqs + (size_t)tileM * KKSTEPS * 1024 + lane * 32;

    const int8_t* bptr = w8 + (size_t)(colBase + l15) * KN + (hi << 4);

    v8i acc0 = (v8i){0,0,0,0,0,0,0,0};
    v8i acc1 = (v8i){0,0,0,0,0,0,0,0};
    v8i acc2 = (v8i){0,0,0,0,0,0,0,0};
    v8i acc3 = (v8i){0,0,0,0,0,0,0,0};

    #pragma unroll 1
    for (int kk = 0; kk < KKSTEPS; ++kk) {
        const int k0 = kk * 64;
        v4i a0 = *(const v4i*)(aptr + kk * 1024);
        v4i a1 = *(const v4i*)(aptr + kk * 1024 + 16);
        v4i b00 = *(const v4i*)(bptr + 0 * (16 * KN) + k0);
        v4i b01 = *(const v4i*)(bptr + 0 * (16 * KN) + k0 + 32);
        v4i b10 = *(const v4i*)(bptr + 1 * (16 * KN) + k0);
        v4i b11 = *(const v4i*)(bptr + 1 * (16 * KN) + k0 + 32);
        v4i b20 = *(const v4i*)(bptr + 2 * (16 * KN) + k0);
        v4i b21 = *(const v4i*)(bptr + 2 * (16 * KN) + k0 + 32);
        v4i b30 = *(const v4i*)(bptr + 3 * (16 * KN) + k0);
        v4i b31 = *(const v4i*)(bptr + 3 * (16 * KN) + k0 + 32);

        const v8i A = pack8(a0, a1);
        acc0 = __builtin_amdgcn_wmma_i32_16x16x64_iu8(
            true, A, true, pack8(b00, b01), acc0, false, false);
        acc1 = __builtin_amdgcn_wmma_i32_16x16x64_iu8(
            true, A, true, pack8(b10, b11), acc1, false, false);
        acc2 = __builtin_amdgcn_wmma_i32_16x16x64_iu8(
            true, A, true, pack8(b20, b21), acc2, false, false);
        acc3 = __builtin_amdgcn_wmma_i32_16x16x64_iu8(
            true, A, true, pack8(b30, b31), acc3, false, false);
    }

    const int rowA = tileM * 16 + l15;
    v8f facc0 = (v8f){0.f,0.f,0.f,0.f,0.f,0.f,0.f,0.f};
    v8f facc1 = (v8f){0.f,0.f,0.f,0.f,0.f,0.f,0.f,0.f};
    v8f facc2 = (v8f){0.f,0.f,0.f,0.f,0.f,0.f,0.f,0.f};
    v8f facc3 = (v8f){0.f,0.f,0.f,0.f,0.f,0.f,0.f,0.f};

    {
        const float* apo = xo + (size_t)rowA * KO + (hi << 1);
        const float* bpo = wo + (size_t)(colBase + l15) * KO + (hi << 1);
        #pragma unroll 1
        for (int kk = 0; kk < KO / 4; ++kk) {
            v2f af  = *(const v2f*)(apo + kk * 4);
            v2f bf0 = *(const v2f*)(bpo + 0 * (16 * KO) + kk * 4);
            v2f bf1 = *(const v2f*)(bpo + 1 * (16 * KO) + kk * 4);
            v2f bf2 = *(const v2f*)(bpo + 2 * (16 * KO) + kk * 4);
            v2f bf3 = *(const v2f*)(bpo + 3 * (16 * KO) + kk * 4);
            facc0 = __builtin_amdgcn_wmma_f32_16x16x4_f32(
                false, af, false, bf0, (short)0, facc0, false, false);
            facc1 = __builtin_amdgcn_wmma_f32_16x16x4_f32(
                false, af, false, bf1, (short)0, facc1, false, false);
            facc2 = __builtin_amdgcn_wmma_f32_16x16x4_f32(
                false, af, false, bf2, (short)0, facc2, false, false);
            facc3 = __builtin_amdgcn_wmma_f32_16x16x4_f32(
                false, af, false, bf3, (short)0, facc3, false, false);
        }
    }

    const int rbase = tileM * 16 + (hi << 3);
    float xsv[8];
    #pragma unroll
    for (int g = 0; g < 8; ++g) xsv[g] = xs[rbase + g];

    const int c0 = colBase + l15;
    const float w0 = wsc[c0];
    const float w1 = wsc[c0 + 16];
    const float w2 = wsc[c0 + 32];
    const float w3 = wsc[c0 + 48];

    #pragma unroll
    for (int g = 0; g < 8; ++g) {
        const size_t rb = (size_t)(rbase + g) * NDIM + c0;
        out[rb +  0] = (float)acc0[g] * xsv[g] * w0 + facc0[g];
        out[rb + 16] = (float)acc1[g] * xsv[g] * w1 + facc1[g];
        out[rb + 32] = (float)acc2[g] * xsv[g] * w2 + facc2[g];
        out[rb + 48] = (float)acc3[g] * xsv[g] * w3 + facc3[g];
    }
}

extern "C" void kernel_launch(void* const* d_in, const int* in_sizes, int n_in,
                              void* d_out, int out_size, void* d_ws, size_t ws_size,
                              hipStream_t stream) {
    const float*  x    = (const float*)d_in[0];
    const int8_t* w8   = (const int8_t*)d_in[1];
    const float*  wsc  = (const float*)d_in[2];
    const float*  wo   = (const float*)d_in[3];
    const int*    nidx = (const int*)d_in[4];
    const int*    oidx = (const int*)d_in[5];
    float*        out  = (float*)d_out;

    int8_t* xqs = (int8_t*)d_ws + XQ_OFF;
    float*  xo  = (float*)((char*)d_ws + XO_OFF);
    float*  xs  = (float*)((char*)d_ws + XS_OFF);

    quant_gather_kernel<<<BATCH_T, 256, 0, stream>>>(x, nidx, oidx, xqs, xo, xs);

    wmma_gemm_kernel<<<(BATCH_T / 16) * (NDIM / 64) / 8, 256, 0, stream>>>(
        xqs, xs, w8, wsc, xo, wo, out);
}
